// Highway_gate_gcn_84301618085976
// MI455X (gfx1250) — matrix operand order repaired and checked by reading, not compiled, not run on hardware
//
#include <hip/hip_runtime.h>
#include <cstdint>
#include <cstddef>

// ---------------------------------------------------------------------------
// Types for CDNA5 WMMA (wave32): v_wmma_f32_16x16x32_bf16
// ---------------------------------------------------------------------------
typedef __attribute__((ext_vector_type(16))) __bf16 v16bf;
typedef __attribute__((ext_vector_type(8)))  __bf16 v8bf;
typedef __attribute__((ext_vector_type(8)))  float  v8f;

__device__ __forceinline__ unsigned short f2bfbits(float f) {
    unsigned u = __builtin_bit_cast(unsigned, f);
    unsigned r = u + 0x7FFFu + ((u >> 16) & 1u);   // round-to-nearest-even
    return (unsigned short)(r >> 16);
}
__device__ __forceinline__ __bf16 f2bf(float f) {
    unsigned short s = f2bfbits(f);
    return __builtin_bit_cast(__bf16, s);
}

// ---------------------------------------------------------------------------
// Pack a KxN f32 weight matrix into WMMA B-fragment layout (bf16):
//   fragment index: (kt*NTILES + nt) ; per fragment 32 lanes x 16 bf16
//   lane L: column n = nt*16 + (L & 15) ; K = kt*32 + (L>>4)*16 + e  (e=0..15)
// ---------------------------------------------------------------------------
__global__ __launch_bounds__(256) void pack_b_kernel(
    const float* __restrict__ W, unsigned short* __restrict__ P, int K, int N)
{
    int idx = blockIdx.x * blockDim.x + threadIdx.x;
    int total = (K / 32) * (N / 16) * 32;
    if (idx >= total) return;
    int lane   = idx & 31;
    int tile   = idx >> 5;
    int ntiles = N / 16;
    int kt = tile / ntiles;
    int nt = tile % ntiles;
    int n     = nt * 16 + (lane & 15);
    int kbase = kt * 32 + (lane >> 4) * 16;
#pragma unroll
    for (int e = 0; e < 16; ++e) {
        float f = W[(size_t)(kbase + e) * N + n];
        P[(size_t)idx * 16 + e] = f2bfbits(f);
    }
}

// ---------------------------------------------------------------------------
// GEMM: C[M x N] = A[M x KDIM] @ Bpacked ; N = NTILES*16.
// A is f32 (on-the-fly bf16 cvt) when AF32, else pre-converted bf16.
// Block = 256 threads = 8 waves; each wave owns a 16-row strip and all NTILES
// column tiles. Packed B k-slabs stream through a double-buffered LDS tile
// (global loads for slab kt+1 issue before the WMMA chain over slab kt).
// The WMMA chain itself runs a 2-deep software pipeline over B fragments so
// ds_load latency overlaps WMMA issue instead of serializing on dscnt==0.
// EXEC stays all-ones at every WMMA (row clamping + store guards, no early
// return — barriers require all waves resident).
// ---------------------------------------------------------------------------
template <int NTILES, int KDIM, bool AF32>
__global__ __launch_bounds__(256) void gemm_wmma(
    const void* __restrict__ Av, const unsigned short* __restrict__ Bp,
    float* __restrict__ C, int M)
{
    constexpr int N        = NTILES * 16;
    constexpr int KT       = KDIM / 32;
    constexpr int TILE_USH = NTILES * 32 * 16;               // ushorts per k-slab
    constexpr int CHUNKS   = (TILE_USH * 2) / (256 * 16);    // uint4s per thread
    __shared__ unsigned short ldsB[2][TILE_USH];

    const int tid  = threadIdx.x;
    const int lane = tid & 31;
    const int wave = tid >> 5;
    const int row0 = (blockIdx.x * 8 + wave) * 16;
    const bool waveActive = (row0 < M);

    const int r  = lane & 15;
    const int kg = lane >> 4;
    int arow = row0 + r;
    if (arow >= M) arow = M - 1;    // clamp: garbage rows are store-guarded

    v8f acc[NTILES];
#pragma unroll
    for (int t = 0; t < NTILES; ++t) {
        v8f z = {0.f, 0.f, 0.f, 0.f, 0.f, 0.f, 0.f, 0.f};
        acc[t] = z;
    }

    // ---- prologue: stage k-slab 0 into LDS buffer 0 ----
    {
        const uint4* src = reinterpret_cast<const uint4*>(Bp);
        uint4*       dst = reinterpret_cast<uint4*>(&ldsB[0][0]);
#pragma unroll
        for (int c = 0; c < CHUNKS; ++c) dst[tid + 256 * c] = src[tid + 256 * c];
    }
    __syncthreads();

    for (int kt = 0; kt < KT; ++kt) {
        const int  buf     = kt & 1;
        const bool hasNext = (kt + 1 < KT);

        // 1) issue global loads for the next k-slab (latency hidden by WMMAs)
        uint4 stage[CHUNKS];
        if (hasNext) {
            const uint4* src =
                reinterpret_cast<const uint4*>(Bp + (size_t)(kt + 1) * TILE_USH);
#pragma unroll
            for (int c = 0; c < CHUNKS; ++c) stage[c] = src[tid + 256 * c];
        }

        // 2) A fragment (ISA 7.12.2, 16-bit A 16x32):
        //    lane L: row = strip + (L&15); e0..15 = K kg*16.. (contiguous, matches packed B)
        v16bf afrag;
        if constexpr (AF32) {
            const float* ap = (const float*)Av + (size_t)arow * KDIM + kt * 32 + kg * 16;
            if (hasNext) __builtin_prefetch(ap + 32, 0, 0);
            float4 a0 = *reinterpret_cast<const float4*>(ap);
            float4 a1 = *reinterpret_cast<const float4*>(ap + 4);
            float4 a2 = *reinterpret_cast<const float4*>(ap + 8);
            float4 a3 = *reinterpret_cast<const float4*>(ap + 12);
            afrag[0]  = f2bf(a0.x); afrag[1]  = f2bf(a0.y);
            afrag[2]  = f2bf(a0.z); afrag[3]  = f2bf(a0.w);
            afrag[4]  = f2bf(a1.x); afrag[5]  = f2bf(a1.y);
            afrag[6]  = f2bf(a1.z); afrag[7]  = f2bf(a1.w);
            afrag[8]  = f2bf(a2.x); afrag[9]  = f2bf(a2.y);
            afrag[10] = f2bf(a2.z); afrag[11] = f2bf(a2.w);
            afrag[12] = f2bf(a3.x); afrag[13] = f2bf(a3.y);
            afrag[14] = f2bf(a3.z); afrag[15] = f2bf(a3.w);
        } else {
            const unsigned short* ap =
                (const unsigned short*)Av + (size_t)arow * KDIM + kt * 32 + kg * 16;
            if (hasNext) __builtin_prefetch(ap + 32, 0, 0);
            v8bf lo = __builtin_bit_cast(v8bf, *reinterpret_cast<const uint4*>(ap));
            v8bf hi = __builtin_bit_cast(v8bf, *reinterpret_cast<const uint4*>(ap + 8));
            afrag = __builtin_shufflevector(lo, hi, 0, 1, 2, 3, 4, 5, 6, 7,
                                            8, 9, 10, 11, 12, 13, 14, 15);
        }

        // 3) WMMA chain over the resident LDS slab, 2-deep pipelined B loads:
        //    fragment nt+2 is fetched before WMMA nt issues, keeping two
        //    ds_load pairs in flight under each WMMA.
        const unsigned short* lb = &ldsB[buf][lane * 16];
        v16bf bA = *reinterpret_cast<const v16bf*>(lb);
        v16bf bB = bA;
        if constexpr (NTILES > 1)
            bB = *reinterpret_cast<const v16bf*>(lb + 512);
#pragma unroll
        for (int nt = 0; nt < NTILES; ++nt) {
            v16bf bC = bB;
            if (nt + 2 < NTILES)
                bC = *reinterpret_cast<const v16bf*>(lb + (nt + 2) * 512);
            acc[nt] = __builtin_amdgcn_wmma_f32_16x16x32_bf16(
                /*neg_a=*/false, afrag, /*neg_b=*/false, bA,
                /*c_mod=*/(short)0, acc[nt], /*reuse_a=*/false, /*reuse_b=*/false);
            bA = bB;
            bB = bC;
        }

        // 4) commit next slab to the other LDS buffer
        if (hasNext) {
            uint4* dst = reinterpret_cast<uint4*>(&ldsB[buf ^ 1][0]);
#pragma unroll
            for (int c = 0; c < CHUNKS; ++c) dst[tid + 256 * c] = stage[c];
        }
        __syncthreads();
    }

    // Store: v8f element i -> row row0 + kg*8 + i, col nt*16 + (lane&15)
    if (waveActive) {
        const int col0  = lane & 15;
        const int mbase = row0 + kg * 8;
#pragma unroll
        for (int nt = 0; nt < NTILES; ++nt) {
            int col = nt * 16 + col0;
#pragma unroll
            for (int i = 0; i < 8; ++i) {
                int rr = mbase + i;
                if (rr < M) C[(size_t)rr * N + col] = acc[nt][i];
            }
        }
    }
}

// ---------------------------------------------------------------------------
// SpMM scatter: one wave per edge; 32 lanes cover F contiguous features.
// dst[row, :] += val * src[col, :]   (atomic f32 adds into L2-resident acc)
// ---------------------------------------------------------------------------
template <int F>
__global__ __launch_bounds__(256) void spmm_scatter(
    const int* __restrict__ er, const int* __restrict__ ec,
    const float* __restrict__ ev, const float* __restrict__ src,
    float* __restrict__ dst, int nE)
{
    int wid = blockIdx.x * (blockDim.x >> 5) + (threadIdx.x >> 5);
    if (wid >= nE) return;
    const int lane = threadIdx.x & 31;
    const int rrow = er[wid];
    const int ccol = ec[wid];
    const float v  = ev[wid];
    constexpr int PER = F / 32;
    const float* s = src + (size_t)ccol * F + lane * PER;
    float*       d = dst + (size_t)rrow * F + lane * PER;
    if constexpr (PER >= 4) {
#pragma unroll
        for (int i = 0; i < PER; i += 4) {
            float4 t = *reinterpret_cast<const float4*>(s + i);
            atomicAdd(d + i + 0, v * t.x);
            atomicAdd(d + i + 1, v * t.y);
            atomicAdd(d + i + 2, v * t.z);
            atomicAdd(d + i + 3, v * t.w);
        }
    } else {
#pragma unroll
        for (int i = 0; i < PER; ++i) atomicAdd(d + i, v * s[i]);
    }
}

// ---------------------------------------------------------------------------
// Elementwise helpers
// ---------------------------------------------------------------------------
__global__ __launch_bounds__(256) void zero_f32(float* __restrict__ p, size_t n)
{
    size_t i = (size_t)blockIdx.x * blockDim.x + threadIdx.x;
    size_t stride = (size_t)gridDim.x * blockDim.x;
    for (; i < n; i += stride) p[i] = 0.f;
}

// ACT: 0 = tanh, 1 = sigmoid ; in-place x = act(x + bias[i % F]);
// optionally also emits a bf16 copy (for GEMM A operands).
template <int ACT, bool WBF>
__global__ __launch_bounds__(256) void bias_act(
    float* __restrict__ x, const float* __restrict__ b,
    unsigned short* __restrict__ xbf, int F, size_t n)
{
    size_t i = (size_t)blockIdx.x * blockDim.x + threadIdx.x;
    if (i >= n) return;
    float v = x[i] + b[i & (size_t)(F - 1)];
    float r = (ACT == 0) ? tanhf(v) : 1.f / (1.f + __expf(-v));
    x[i] = r;
    if constexpr (WBF) xbf[i] = f2bfbits(r);
}

// gated = g*h2 + (1-g)*h1 ; emitted directly as bf16 (only GEMM4 consumes it)
__global__ __launch_bounds__(256) void gate_kernel(
    const float* __restrict__ g, const float* __restrict__ h2,
    const float* __restrict__ h1, unsigned short* __restrict__ outbf, size_t n)
{
    size_t i = (size_t)blockIdx.x * blockDim.x + threadIdx.x;
    if (i >= n) return;
    float gg = g[i];
    outbf[i] = f2bfbits(gg * h2[i] + (1.f - gg) * h1[i]);
}

// out[row,:] = log_softmax(acc[row,:] + b)  over 64 classes
__global__ __launch_bounds__(256) void logsoftmax64(
    const float* __restrict__ acc, const float* __restrict__ b,
    float* __restrict__ out, int M)
{
    int row = blockIdx.x * blockDim.x + threadIdx.x;
    if (row >= M) return;
    float v[64];
    float mx = -3.4e38f;
#pragma unroll
    for (int i = 0; i < 64; ++i) {
        v[i] = acc[(size_t)row * 64 + i] + b[i];
        mx = fmaxf(mx, v[i]);
    }
    float s = 0.f;
#pragma unroll
    for (int i = 0; i < 64; ++i) s += __expf(v[i] - mx);
    float ls = mx + __logf(s);
#pragma unroll
    for (int i = 0; i < 64; ++i) out[(size_t)row * 64 + i] = v[i] - ls;
}

// ---------------------------------------------------------------------------
// Host-side orchestration (graph-capture safe: kernels only, all on stream)
// ---------------------------------------------------------------------------
extern "C" void kernel_launch(void* const* d_in, const int* in_sizes, int n_in,
                              void* d_out, int out_size, void* d_ws, size_t ws_size,
                              hipStream_t stream)
{
    (void)n_in; (void)out_size; (void)ws_size;
    const float* x   = (const float*)d_in[0];
    const int*   er  = (const int*)  d_in[1];
    const int*   ec  = (const int*)  d_in[2];
    const float* ev  = (const float*)d_in[3];
    const float* w1  = (const float*)d_in[4];
    const float* b1  = (const float*)d_in[5];
    const float* w2  = (const float*)d_in[6];
    const float* b2  = (const float*)d_in[7];
    const float* fcw = (const float*)d_in[8];
    const float* fcb = (const float*)d_in[9];
    const float* w3  = (const float*)d_in[10];
    const float* b3  = (const float*)d_in[11];
    float* out = (float*)d_out;

    const int M  = in_sizes[0] / 512;   // 100000
    const int nE = in_sizes[1];         // 3200000

    // Workspace carve-up
    float* ws = (float*)d_ws;
    size_t o = 0;
    float* xw1   = ws + o; o += (size_t)M * 256;   // GEMM1 out, later reused as h2 acc
    float* h1    = ws + o; o += (size_t)M * 256;   // spmm1 acc -> tanh in place
    float* t2    = ws + o; o += (size_t)M * 256;   // GEMM2 out
    float* h3    = ws + o; o += (size_t)M * 256;   // GEMM3 out -> sigmoid in place
    float* gw3   = ws + o; o += (size_t)M * 64;    // GEMM4 out
    float* logit = ws + o; o += (size_t)M * 64;    // spmm3 acc
    unsigned short* h1bf  = (unsigned short*)(ws + o); o += (size_t)M * 128; // M*256 ush
    unsigned short* gatbf = (unsigned short*)(ws + o); o += (size_t)M * 128; // M*256 ush
    unsigned short* w1p   = (unsigned short*)(ws + o);
    unsigned short* w2p   = w1p + (size_t)512 * 256;
    unsigned short* fcwp  = w2p + (size_t)256 * 256;
    unsigned short* w3p   = fcwp + (size_t)256 * 256;

    const size_t n256 = (size_t)M * 256;
    const size_t n64  = (size_t)M * 64;
    const int gemmBlocks = (M + 127) / 128;
    const int spmmBlocks = (nE + 7) / 8;
    const int ewBlocks   = (int)((n256 + 255) / 256);

    // 0) pack weights into WMMA B-fragment bf16 layout
    pack_b_kernel<<<(16 * 16 * 32 + 255) / 256, 256, 0, stream>>>(w1, w1p, 512, 256);
    pack_b_kernel<<<( 8 * 16 * 32 + 255) / 256, 256, 0, stream>>>(w2, w2p, 256, 256);
    pack_b_kernel<<<( 8 * 16 * 32 + 255) / 256, 256, 0, stream>>>(fcw, fcwp, 256, 256);
    pack_b_kernel<<<( 8 *  4 * 32 + 255) / 256, 256, 0, stream>>>(w3, w3p, 256, 64);

    // 1) xw1 = x @ w1   (A = f32, converted on the fly — x is read exactly once)
    gemm_wmma<16, 512, true><<<gemmBlocks, 256, 0, stream>>>(x, w1p, xw1, M);

    // 2) h1 = tanh(spmm(adj, xw1) + b1)   (+ bf16 copy for GEMM2/3)
    zero_f32<<<2048, 256, 0, stream>>>(h1, n256);
    spmm_scatter<256><<<spmmBlocks, 256, 0, stream>>>(er, ec, ev, xw1, h1, nE);
    bias_act<0, true><<<ewBlocks, 256, 0, stream>>>(h1, b1, h1bf, 256, n256);

    // 3) t2 = h1 @ w2 ;  h3 = h1 @ fc_w   (A = bf16)
    gemm_wmma<16, 256, false><<<gemmBlocks, 256, 0, stream>>>(h1bf, w2p, t2, M);
    gemm_wmma<16, 256, false><<<gemmBlocks, 256, 0, stream>>>(h1bf, fcwp, h3, M);

    // 4) h2 = sigmoid(spmm(adj, t2) + b2)   (accumulate into reused xw1 buffer)
    zero_f32<<<2048, 256, 0, stream>>>(xw1, n256);
    spmm_scatter<256><<<spmmBlocks, 256, 0, stream>>>(er, ec, ev, t2, xw1, nE);
    bias_act<1, false><<<ewBlocks, 256, 0, stream>>>(xw1, b2, nullptr, 256, n256);

    // 5) h3 = sigmoid(h3 + fc_b)
    bias_act<1, false><<<ewBlocks, 256, 0, stream>>>(h3, fcb, nullptr, 256, n256);

    // 6) gated = h3*h2 + (1-h3)*h1  -> bf16 (only GEMM4 consumes it)
    gate_kernel<<<ewBlocks, 256, 0, stream>>>(h3, xw1, h1, gatbf, n256);

    // 7) gw3 = gated @ w3  (N = 64, A = bf16)
    gemm_wmma<4, 256, false><<<gemmBlocks, 256, 0, stream>>>(gatbf, w3p, gw3, M);

    // 8) logits = spmm(adj, gw3) + b3 ; log_softmax -> out
    zero_f32<<<2048, 256, 0, stream>>>(logit, n64);
    spmm_scatter<64><<<spmmBlocks, 256, 0, stream>>>(er, ec, ev, gw3, logit, nE);
    logsoftmax64<<<(M + 255) / 256, 256, 0, stream>>>(logit, b3, out, M);
}
